// attn_block_separate_66357244723396
// MI455X (gfx1250) — matrix operand order repaired and checked by reading, not compiled, not run on hardware
//
#include <hip/hip_runtime.h>
#include <hip/hip_bf16.h>

typedef __attribute__((ext_vector_type(16))) __bf16 v16bf;
typedef __attribute__((ext_vector_type(8)))  __bf16 v8bf;
typedef __attribute__((ext_vector_type(8)))  float  v8f;
typedef __attribute__((ext_vector_type(4)))  unsigned int v4u;
typedef __attribute__((ext_vector_type(8)))  int v8i;
typedef __attribute__((ext_vector_type(4)))  int v4i;

static constexpr int Bn = 8;
static constexpr int Tn = 2048;
static constexpr int Dn = 512;

__device__ __forceinline__ __bf16 f32_to_bf16(float f) {
  union { float f; unsigned u; } a; a.f = f;
  unsigned r = a.u + 0x7FFFu + ((a.u >> 16) & 1u);
  unsigned short h = (unsigned short)(r >> 16);
  union { unsigned short s; __bf16 b; } c; c.s = h;
  return c.b;
}

__device__ __forceinline__ v16bf concat8(v8bf lo, v8bf hi) {
  return __builtin_shufflevector(lo, hi, 0, 1, 2, 3, 4, 5, 6, 7,
                                 8, 9, 10, 11, 12, 13, 14, 15);
}

__device__ __forceinline__ float dot512(const float* __restrict__ x,
                                        const float* __restrict__ w) {
  float s = 0.f;
  #pragma unroll 8
  for (int i = 0; i < 512; i += 4) {
    float4 a = *(const float4*)(x + i);
    float4 b = *(const float4*)(w + i);
    s = fmaf(a.x, b.x, fmaf(a.y, b.y, fmaf(a.z, b.z, fmaf(a.w, b.w, s))));
  }
  return s;
}

// ---------------------------------------------------------------------------
// Kernel 1: gates from aux_emb (tiny: B=8 rows). One block per batch.
// ---------------------------------------------------------------------------
__device__ void gate_inner(const float* __restrict__ xsh,
                           const float* __restrict__ W,
                           const float* __restrict__ bias, float a,
                           const float* __restrict__ g,
                           const float* __restrict__ beta,
                           float* __restrict__ red,
                           float* __restrict__ out_sh) {
  const int tid = threadIdx.x;
  if (tid == 0) { red[0] = 0.f; red[1] = 0.f; }
  __syncthreads();
  const int j0 = tid, j1 = tid + 256;
  float h0 = dot512(xsh, W + (size_t)j0 * 512) + bias[j0];
  float h1 = dot512(xsh, W + (size_t)j1 * 512) + bias[j1];
  h0 = (h0 >= 0.f) ? h0 : a * h0;
  h1 = (h1 >= 0.f) ? h1 : a * h1;
  atomicAdd(&red[0], h0 + h1);
  atomicAdd(&red[1], h0 * h0 + h1 * h1);
  __syncthreads();
  const float mu   = red[0] * (1.f / 512.f);
  const float var  = red[1] * (1.f / 512.f) - mu * mu;
  const float rstd = rsqrtf(var + 1e-5f);
  const float y0 = (h0 - mu) * rstd * g[j0] + beta[j0];
  const float y1 = (h1 - mu) * rstd * g[j1] + beta[j1];
  out_sh[j0] = 1.f / (1.f + __expf(-y0));
  out_sh[j1] = 1.f / (1.f + __expf(-y1));
  __syncthreads();
}

__global__ __launch_bounds__(256)
void gates_kernel(const float* __restrict__ aux,
                  const float* __restrict__ Wqg, const float* __restrict__ bqg,
                  const float* __restrict__ aq,  const float* __restrict__ gqv,
                  const float* __restrict__ betaq,
                  const float* __restrict__ Wkg, const float* __restrict__ bkg,
                  const float* __restrict__ ak,  const float* __restrict__ gkv,
                  const float* __restrict__ betak,
                  const float* __restrict__ Wvg, const float* __restrict__ bvg,
                  const float* __restrict__ av,  const float* __restrict__ gvv,
                  const float* __restrict__ betav,
                  const float* __restrict__ Wsig, const float* __restrict__ bsig,
                  const float* __restrict__ Wtan, const float* __restrict__ btan,
                  float* __restrict__ gq_out, float* __restrict__ gk_out,
                  float* __restrict__ gv_out) {
  __shared__ __align__(16) float xsh[512];
  __shared__ float red[2];
  __shared__ __align__(16) float gsh[512];
  const int b = blockIdx.x;
  const int tid = threadIdx.x;
  xsh[tid]       = aux[b * 512 + tid];
  xsh[tid + 256] = aux[b * 512 + tid + 256];
  __syncthreads();

  gate_inner(xsh, Wqg, bqg, aq[0], gqv, betaq, red, gsh);
  gq_out[b * 512 + tid]       = gsh[tid];
  gq_out[b * 512 + tid + 256] = gsh[tid + 256];
  __syncthreads();

  gate_inner(xsh, Wkg, bkg, ak[0], gkv, betak, red, gsh);
  gk_out[b * 512 + tid]       = gsh[tid];
  gk_out[b * 512 + tid + 256] = gsh[tid + 256];
  __syncthreads();

  gate_inner(xsh, Wvg, bvg, av[0], gvv, betav, red, gsh);
  // vg = sigmoid(vg@Wsig.T + bsig) * tanh(vg@Wtan.T + btan)
  #pragma unroll
  for (int p = 0; p < 2; ++p) {
    const int j = tid + p * 256;
    const float s = dot512(gsh, Wsig + (size_t)j * 512) + bsig[j];
    const float t = dot512(gsh, Wtan + (size_t)j * 512) + btan[j];
    gv_out[b * 512 + j] = (1.f / (1.f + __expf(-s))) * tanhf(t);
  }
}

// ---------------------------------------------------------------------------
// Kernel 2: Wq (512x512 f32) -> bf16
// ---------------------------------------------------------------------------
__global__ __launch_bounds__(256)
void wq_to_bf16_kernel(const float* __restrict__ Wq, __bf16* __restrict__ out) {
  const int i = (blockIdx.x * 256 + threadIdx.x) * 4;
  float4 v = *(const float4*)(Wq + i);
  out[i + 0] = f32_to_bf16(v.x);
  out[i + 1] = f32_to_bf16(v.y);
  out[i + 2] = f32_to_bf16(v.z);
  out[i + 3] = f32_to_bf16(v.w);
}

// ---------------------------------------------------------------------------
// Kernel 3: q = (query @ Wq^T + bq) * gate_q  -> bf16  (WMMA GEMM)
// grid: (BT/128, D/64), 256 threads = 8 waves; wave w owns 16-row strip.
// ---------------------------------------------------------------------------
__global__ __launch_bounds__(256)
void qproj_kernel(const float* __restrict__ query,
                  const __bf16* __restrict__ Wqbf,
                  const float* __restrict__ bq,
                  const float* __restrict__ gq_ws,
                  __bf16* __restrict__ qbf) {
  __shared__ __align__(16) __bf16 At[128 * 32];
  const int tid  = threadIdx.x;
  const int lane = tid & 31;
  const int ln   = lane & 15;
  const int hi   = lane >> 4;
  const int strip = tid >> 5;
  const int rb = blockIdx.x * 128;
  const int cb = blockIdx.y * 64;

  v8f zero8 = {};
  v8f acc[4];
  #pragma unroll
  for (int t = 0; t < 4; ++t) acc[t] = zero8;

  const int srow = tid >> 1;
  const int scol = (tid & 1) * 16;

  for (int kc = 0; kc < 16; ++kc) {
    const int k0 = kc * 32;
    {  // stage A tile (128x32) as bf16 in LDS
      const float* src = query + ((size_t)(rb + srow)) * Dn + k0 + scol;
      float4 f0 = *(const float4*)(src + 0);
      float4 f1 = *(const float4*)(src + 4);
      float4 f2 = *(const float4*)(src + 8);
      float4 f3 = *(const float4*)(src + 12);
      __bf16* dst = &At[srow * 32 + scol];
      dst[0] = f32_to_bf16(f0.x);  dst[1] = f32_to_bf16(f0.y);
      dst[2] = f32_to_bf16(f0.z);  dst[3] = f32_to_bf16(f0.w);
      dst[4] = f32_to_bf16(f1.x);  dst[5] = f32_to_bf16(f1.y);
      dst[6] = f32_to_bf16(f1.z);  dst[7] = f32_to_bf16(f1.w);
      dst[8] = f32_to_bf16(f2.x);  dst[9] = f32_to_bf16(f2.y);
      dst[10] = f32_to_bf16(f2.z); dst[11] = f32_to_bf16(f2.w);
      dst[12] = f32_to_bf16(f3.x); dst[13] = f32_to_bf16(f3.y);
      dst[14] = f32_to_bf16(f3.z); dst[15] = f32_to_bf16(f3.w);
    }
    __syncthreads();
    // A fragment: 16x32 bf16, rows strip*16..+15
    const __bf16* arow = &At[(strip * 16 + ln) * 32];
    v16bf af = concat8(*(const v8bf*)(arow + hi * 16),
                       *(const v8bf*)(arow + hi * 16 + 8));
    #pragma unroll
    for (int t = 0; t < 4; ++t) {
      const int n = cb + t * 16 + ln;  // B col = Wq row (B = Wq^T)
      const __bf16* brow = Wqbf + (size_t)n * Dn + k0 + hi * 16;
      v16bf bf = concat8(*(const v8bf*)(brow), *(const v8bf*)(brow + 8));
      if (t == 0)
        acc[t] = __builtin_amdgcn_wmma_f32_16x16x32_bf16(
            false, af, false, bf, (short)0, acc[t], false, false);
      else
        acc[t] = __builtin_amdgcn_wmma_f32_16x16x32_bf16(
            false, af, false, bf, (short)0, acc[t], true, false);
    }
    __syncthreads();
  }
  // epilogue: + bq, * gate_q[batch], -> bf16
  #pragma unroll
  for (int t = 0; t < 4; ++t) {
    const int col = cb + t * 16 + ln;
    const float bqv = bq[col];
    #pragma unroll
    for (int j = 0; j < 8; ++j) {
      const int row = rb + strip * 16 + j + hi * 8;
      const int bb  = row >> 11;  // /T
      float v = acc[t][j] + bqv;
      v *= gq_ws[bb * Dn + col];
      qbf[(size_t)row * Dn + col] = f32_to_bf16(v);
    }
  }
}

// ---------------------------------------------------------------------------
// Kernel 4: k = value*gate_k -> bf16 ; v = value*gate_v -> bf16
// ---------------------------------------------------------------------------
__global__ __launch_bounds__(256)
void kv_gate_kernel(const float* __restrict__ value,
                    const float* __restrict__ gk_ws,
                    const float* __restrict__ gv_ws,
                    __bf16* __restrict__ kbf, __bf16* __restrict__ vbf) {
  const size_t i = ((size_t)blockIdx.x * 256 + threadIdx.x) * 4;
  float4 v = *(const float4*)(value + i);
  const int d0 = (int)(i & (size_t)(Dn - 1));
  const int b  = (int)(i >> 20);  // / (T*D) = 2^20
  float4 g4k = *(const float4*)(gk_ws + b * Dn + d0);
  float4 g4v = *(const float4*)(gv_ws + b * Dn + d0);
  kbf[i + 0] = f32_to_bf16(v.x * g4k.x);
  kbf[i + 1] = f32_to_bf16(v.y * g4k.y);
  kbf[i + 2] = f32_to_bf16(v.z * g4k.z);
  kbf[i + 3] = f32_to_bf16(v.w * g4k.w);
  vbf[i + 0] = f32_to_bf16(v.x * g4v.x);
  vbf[i + 1] = f32_to_bf16(v.y * g4v.y);
  vbf[i + 2] = f32_to_bf16(v.z * g4v.z);
  vbf[i + 3] = f32_to_bf16(v.w * g4v.w);
}

// ---------------------------------------------------------------------------
// Kernel 5: causal + length-masked flash attention, bf16 WMMA, f32 softmax.
// grid (T/64, B); 256 threads = 8 waves; wave (mw,dw): M-strip mw (16 rows),
// D-half dw (256 cols). BLOCK_N = 32 keys per iteration.
// K tile staged via the Tensor Data Mover (async, TENSORcnt); V tile staged
// manually because it must be transposed for the PV B-matrix.
// ---------------------------------------------------------------------------
__global__ __launch_bounds__(256)
void flash_kernel(const __bf16* __restrict__ qbf,
                  const __bf16* __restrict__ kbf,
                  const __bf16* __restrict__ vbf,
                  const int* __restrict__ input_len,
                  float* __restrict__ out) {
  constexpr int BN = 32;
  __shared__ __align__(16) __bf16 Kt[BN * 512];   // 32 KB (TDM destination)
  __shared__ __align__(16) __bf16 VT[512 * BN];   // 32 KB (transposed)
  __shared__ float Sbuf[2][64 * BN];              // 16 KB
  __shared__ __align__(16) __bf16 Pbuf[64 * BN];  // 4 KB
  __shared__ float m_s[64], l_s[64], alpha_s[64];

  const int b    = blockIdx.y;
  const int mb   = blockIdx.x;
  const int row0 = mb * 64;
  const int tid  = threadIdx.x;
  const int lane = tid & 31;
  const int w    = tid >> 5;
  const int mw   = w & 3;
  const int dw   = w >> 2;
  const int ln   = lane & 15;
  const int hi   = lane >> 4;
  const int L    = input_len[b];

  if (tid < 64) { m_s[tid] = -1e30f; l_s[tid] = 0.f; alpha_s[tid] = 0.f; }

  // Q fragments for this wave: rows row0+mw*16+ln, D-half dw (kept in VGPRs)
  v16bf qf[8];
  {
    const size_t qrow = ((size_t)b * Tn + row0 + mw * 16 + ln) * Dn;
    #pragma unroll
    for (int kc = 0; kc < 8; ++kc) {
      const int d0 = dw * 256 + kc * 32;
      qf[kc] = concat8(*(const v8bf*)(qbf + qrow + d0 + hi * 16),
                       *(const v8bf*)(qbf + qrow + d0 + hi * 16 + 8));
    }
  }

  v8f zero8 = {};
  v8f oacc[16];
  #pragma unroll
  for (int t = 0; t < 16; ++t) oacc[t] = zero8;

  __syncthreads();

  const unsigned lds_kt = (unsigned)(unsigned long long)(const void*)&Kt[0];

  const int nb_end = 2 * mb + 2;  // causal: keys up to row0+63
  for (int nb = 0; nb < nb_end; ++nb) {
    const int s0 = nb * BN;
    const size_t kvbase = ((size_t)b * Tn + s0) * Dn;

    // --- async TDM load of the K tile (32 rows x 1024 B) into LDS ---------
    if (w == 0) {
      const unsigned long long ga =
          (unsigned long long)(const void*)(kbf + kvbase);
      // D# group0: count=1 | lds_addr | global_addr(57b) | type=2
      v4u g0 = {1u, lds_kt, (unsigned)(ga & 0xffffffffu),
                (unsigned)((ga >> 32) & 0x01ffffffu) | (2u << 30)};
      // D# group1: data_size=4B; tensor_dim0=256 dw; tensor_dim1=32;
      // tile_dim0=256; tile_dim1=32; tensor_dim0_stride=256 dw.
      v8i g1 = {(int)(2u << 16),    // workgroup_mask=0, data_size=2 (4B)
                (int)(256u << 16),  // tensor_dim0[15:0] in bits 63:48
                (int)(32u << 16),   // tensor_dim0[31:16]=0 | tensor_dim1 lo
                (int)(256u << 16),  // tensor_dim1 hi=0 | tile_dim0=256
                32,                 // tile_dim1=32 | tile_dim2=0
                256,                // tensor_dim0_stride[31:0]
                0, 0};              // stride0 hi, tensor_dim1_stride
      v4i gz4 = {0, 0, 0, 0};
      v8i gz8 = {0, 0, 0, 0, 0, 0, 0, 0};
      __builtin_amdgcn_tensor_load_to_lds(g0, g1, gz4, gz4, gz8, 0);
    }

    // --- manual transposed V staging (pack 2 rows per b32 store) ----------
    if (nb + 1 < nb_end)
      __builtin_prefetch(vbf + kvbase + (size_t)BN * Dn + tid * 8, 0, 1);
    #pragma unroll
    for (int it = 0; it < 4; ++it) {
      const int task = tid + it * 256;   // 1024 tasks: 16 row-pairs x 64 cols
      const int rp   = task >> 6;        // row pair 0..15
      const int c    = (task & 63) * 8;  // col chunk
      union { uint4 u4; unsigned short s[8]; } a0, a1;
      a0.u4 = *(const uint4*)(vbf + kvbase + (size_t)(2 * rp) * Dn + c);
      a1.u4 = *(const uint4*)(vbf + kvbase + (size_t)(2 * rp + 1) * Dn + c);
      #pragma unroll
      for (int j = 0; j < 8; ++j) {
        const unsigned pk = (unsigned)a0.s[j] | ((unsigned)a1.s[j] << 16);
        *(unsigned*)&VT[(c + j) * BN + 2 * rp] = pk;
      }
    }
    if (w == 0) __builtin_amdgcn_s_wait_tensorcnt(0);
    __syncthreads();

    // --- partial S = Q(dw half) @ K^T  -> 16x32 per wave, f32 accum -------
    v8f sacc[2];
    sacc[0] = zero8;
    sacc[1] = zero8;
    #pragma unroll
    for (int kc = 0; kc < 8; ++kc) {
      const int d0 = dw * 256 + kc * 32 + hi * 16;
      const int kk0 = ln;       // key index, tile 0
      const int kk1 = 16 + ln;  // key index, tile 1
      v16bf bf0 = concat8(*(const v8bf*)&Kt[kk0 * 512 + d0],
                          *(const v8bf*)&Kt[kk0 * 512 + d0 + 8]);
      v16bf bf1 = concat8(*(const v8bf*)&Kt[kk1 * 512 + d0],
                          *(const v8bf*)&Kt[kk1 * 512 + d0 + 8]);
      sacc[0] = __builtin_amdgcn_wmma_f32_16x16x32_bf16(
          false, qf[kc], false, bf0, (short)0, sacc[0], false, false);
      sacc[1] = __builtin_amdgcn_wmma_f32_16x16x32_bf16(
          false, qf[kc], false, bf1, (short)0, sacc[1], true, false);
    }
    #pragma unroll
    for (int t = 0; t < 2; ++t)
      #pragma unroll
      for (int j = 0; j < 8; ++j)
        Sbuf[dw][(mw * 16 + j + hi * 8) * BN + t * 16 + ln] = sacc[t][j];
    __syncthreads();

    // --- online softmax update: dw==0 waves own their 16-row strip --------
    if (dw == 0) {
      const int rl = mw * 16 + ln;
      const int tg = row0 + rl;
      float sv[16];
      float mx = -1e30f;
      #pragma unroll
      for (int i = 0; i < 16; ++i) {
        const int c = hi * 16 + i;
        float s = (Sbuf[0][rl * BN + c] + Sbuf[1][rl * BN + c]) *
                  0.04419417382415922f;  // 1/sqrt(512)
        const int sg = s0 + c;
        if (sg > tg || (tg < L && sg >= L)) s = -1e30f;
        sv[i] = s;
        mx = fmaxf(mx, s);
      }
      mx = fmaxf(mx, __shfl_xor(mx, 16, 32));
      const float m_old = m_s[rl];
      const float m_new = fmaxf(m_old, mx);
      const float alpha = __expf(m_old - m_new);
      float sum = 0.f;
      #pragma unroll
      for (int i = 0; i < 16; ++i) {
        const float p = (sv[i] <= -1e29f) ? 0.f : __expf(sv[i] - m_new);
        Pbuf[rl * BN + hi * 16 + i] = f32_to_bf16(p);
        sum += p;
      }
      sum += __shfl_xor(sum, 16, 32);
      if (hi == 0) {
        m_s[rl] = m_new;
        l_s[rl] = l_s[rl] * alpha + sum;
        alpha_s[rl] = alpha;
      }
    }
    __syncthreads();

    // --- rescale running O by alpha ---------------------------------------
    float al[8];
    #pragma unroll
    for (int j = 0; j < 8; ++j) al[j] = alpha_s[mw * 16 + j + hi * 8];
    #pragma unroll
    for (int t = 0; t < 16; ++t)
      #pragma unroll
      for (int j = 0; j < 8; ++j) oacc[t][j] *= al[j];

    // --- O += P @ V  (K = 32 -> one WMMA step per 16-col tile) ------------
    {
      const int pr = mw * 16 + ln;
      v16bf pf = concat8(*(const v8bf*)&Pbuf[pr * BN + hi * 16],
                         *(const v8bf*)&Pbuf[pr * BN + hi * 16 + 8]);
      #pragma unroll
      for (int t = 0; t < 16; ++t) {
        const int d = dw * 256 + t * 16 + ln;  // output column (B-matrix col)
        v16bf bf = concat8(*(const v8bf*)&VT[d * BN + hi * 16],
                           *(const v8bf*)&VT[d * BN + hi * 16 + 8]);
        if (t == 0)
          oacc[t] = __builtin_amdgcn_wmma_f32_16x16x32_bf16(
              false, pf, false, bf, (short)0, oacc[t], false, false);
        else
          oacc[t] = __builtin_amdgcn_wmma_f32_16x16x32_bf16(
              false, pf, false, bf, (short)0, oacc[t], true, false);
      }
    }
    __syncthreads();
  }

  // epilogue: normalize by l, zero rows >= input_len
  #pragma unroll
  for (int j = 0; j < 8; ++j) {
    const int rl = mw * 16 + j + hi * 8;
    const int tg = row0 + rl;
    const float lv = l_s[rl];
    const float linv = (tg < L && lv > 0.f) ? (1.f / lv) : 0.f;
    #pragma unroll
    for (int t = 0; t < 16; ++t) {
      const int col = dw * 256 + t * 16 + ln;
      out[((size_t)b * Tn + tg) * Dn + col] = oacc[t][j] * linv;
    }
  }
}

// ---------------------------------------------------------------------------
extern "C" void kernel_launch(void* const* d_in, const int* in_sizes, int n_in,
                              void* d_out, int out_size, void* d_ws,
                              size_t ws_size, hipStream_t stream) {
  (void)in_sizes; (void)n_in; (void)out_size; (void)ws_size;
  const float* query = (const float*)d_in[0];
  const float* value = (const float*)d_in[1];
  const float* aux   = (const float*)d_in[2];
  const int*   ilen  = (const int*)d_in[3];
  const float* Wq    = (const float*)d_in[4];
  const float* bq    = (const float*)d_in[5];
  const float* Wqg   = (const float*)d_in[6];
  const float* bqg   = (const float*)d_in[7];
  const float* aq    = (const float*)d_in[8];
  const float* gq    = (const float*)d_in[9];
  const float* betaq = (const float*)d_in[10];
  const float* Wkg   = (const float*)d_in[11];
  const float* bkg   = (const float*)d_in[12];
  const float* ak    = (const float*)d_in[13];
  const float* gk    = (const float*)d_in[14];
  const float* betak = (const float*)d_in[15];
  const float* Wvg   = (const float*)d_in[16];
  const float* bvg   = (const float*)d_in[17];
  const float* av    = (const float*)d_in[18];
  const float* gv    = (const float*)d_in[19];
  const float* betav = (const float*)d_in[20];
  const float* Wsig  = (const float*)d_in[21];
  const float* bsig  = (const float*)d_in[22];
  const float* Wtan  = (const float*)d_in[23];
  const float* btan  = (const float*)d_in[24];

  char* ws = (char*)d_ws;
  float* gq_ws = (float*)ws;                 // 8*512 f32
  float* gk_ws = gq_ws + Bn * Dn;
  float* gv_ws = gk_ws + Bn * Dn;
  __bf16* Wqbf = (__bf16*)(ws + 49152);                // 512*512 bf16
  __bf16* qbf  = (__bf16*)(ws + 49152 + 524288);       // B*T*D bf16
  __bf16* kbf  = qbf + (size_t)Bn * Tn * Dn;
  __bf16* vbf  = kbf + (size_t)Bn * Tn * Dn;

  gates_kernel<<<Bn, 256, 0, stream>>>(aux, Wqg, bqg, aq, gq, betaq,
                                       Wkg, bkg, ak, gk, betak,
                                       Wvg, bvg, av, gv, betav,
                                       Wsig, bsig, Wtan, btan,
                                       gq_ws, gk_ws, gv_ws);
  wq_to_bf16_kernel<<<256, 256, 0, stream>>>(Wq, Wqbf);
  qproj_kernel<<<dim3((Bn * Tn) / 128, Dn / 64), 256, 0, stream>>>(
      query, Wqbf, bq, gq_ws, qbf);
  kv_gate_kernel<<<(Bn * Tn * Dn) / (256 * 4), 256, 0, stream>>>(
      value, gk_ws, gv_ws, kbf, vbf);
  flash_kernel<<<dim3(Tn / 64, Bn), 256, 0, stream>>>(
      qbf, kbf, vbf, ilen, (float*)d_out);
}
